// PointCloudEncoder_55825984913689
// MI455X (gfx1250) — matrix operand order repaired and checked by reading, not compiled, not run on hardware
//
#include <hip/hip_runtime.h>
#include <hip/hip_bf16.h>

// ---------------------------------------------------------------------------
// Types for CDNA5 WMMA
// ---------------------------------------------------------------------------
typedef _Float16 half8  __attribute__((ext_vector_type(8)));
typedef _Float16 half16 __attribute__((ext_vector_type(16)));
typedef float    float8 __attribute__((ext_vector_type(8)));
typedef float    f32x2  __attribute__((ext_vector_type(2)));

__device__ __forceinline__ void lds_fence() {
    // wait for all outstanding LDS ops of this wave (CDNA5 split counters)
    asm volatile("s_wait_dscnt 0" ::: "memory");
}

__device__ __forceinline__ half16 load16h(const _Float16* p0, const _Float16* p1) {
    union { half16 v; half8 h[2]; } u;
    u.h[0] = *(const half8*)p0;
    u.h[1] = *(const half8*)p1;
    return u.v;
}

__device__ __forceinline__ float selu_f(float x) {
    const float a = 1.6732632423543772f;
    const float s = 1.0507009873554805f;
    return x > 0.f ? s * x : s * a * (expf(x) - 1.f);
}

// ---------------------------------------------------------------------------
// Weight transpose + fp16 conversion:  Wt[out][k] = (f16) W[k][out], k padded
// ---------------------------------------------------------------------------
__global__ void wcvt_kernel(const float* __restrict__ W, int kin,
                            _Float16* __restrict__ dst, int kpad) {
    int t = blockIdx.x * blockDim.x + threadIdx.x;
    if (t < 64 * kpad) {
        int o = t / kpad, k = t - o * kpad;
        dst[t] = (k < kin) ? (_Float16)W[k * 64 + o] : (_Float16)0.f;
    }
}

// ---------------------------------------------------------------------------
// Scalar kNN (used for CIN=3 layer and tiny bbox graphs)
// ---------------------------------------------------------------------------
template <int CIN, int K, int CHUNK>
__global__ __launch_bounds__(512) void knn_kernel(const float* __restrict__ X,
                                                  int* __restrict__ idxOut, int P) {
    __shared__ float chunk[CHUNK * CIN];
    __shared__ float sqv[CHUNK];
    const int b = blockIdx.x;
    const int i = threadIdx.x;
    const float* Xb = X + (size_t)b * P * CIN;

    float xi[CIN];
    float sqi = 0.f;
    if (i < P) {
#pragma unroll
        for (int c = 0; c < CIN; ++c) { xi[c] = Xb[(size_t)i * CIN + c]; sqi += xi[c] * xi[c]; }
    }
    float best[K]; int bestI[K];
#pragma unroll
    for (int k = 0; k < K; ++k) { best[k] = 3.4e38f; bestI[k] = 0; }

    for (int q0 = 0; q0 < P; q0 += CHUNK) {
        int cnt = (CHUNK < P - q0) ? CHUNK : (P - q0);
        __syncthreads();
        for (int t = threadIdx.x; t < cnt * CIN; t += blockDim.x)
            chunk[t] = Xb[(size_t)q0 * CIN + t];
        __syncthreads();
        for (int t = threadIdx.x; t < cnt; t += blockDim.x) {
            float s = 0.f;
            for (int c = 0; c < CIN; ++c) { float v = chunk[t * CIN + c]; s += v * v; }
            sqv[t] = s;
        }
        __syncthreads();
        if (i < P) {
            for (int j = 0; j < cnt; ++j) {
                float dot = 0.f;
#pragma unroll 4
                for (int c = 0; c < CIN; ++c) dot += xi[c] * chunk[j * CIN + c];
                float d = sqi + sqv[j] - 2.f * dot;
                if (q0 + j == i) d += 1e9f;
                if (d < best[K - 1]) {
                    int k = K - 1;
                    while (k > 0 && best[k - 1] > d) {
                        best[k] = best[k - 1]; bestI[k] = bestI[k - 1]; --k;
                    }
                    best[k] = d; bestI[k] = q0 + j;
                }
            }
        }
    }
    if (i < P) {
        int* op = idxOut + ((size_t)b * P + i) * K;
#pragma unroll
        for (int k = 0; k < K; ++k) op[k] = bestI[k];
    }
}

// ---------------------------------------------------------------------------
// kNN via fp32 WMMA gram matrix (C=64, P=512).
// grid = NOBJ*32 (16-row strips). Block 256 = 8 waves; each wave owns 64 cols.
// G tile via v_wmma_f32_16x16x4_f32; A lane: row m, K=hi*2+{0,1};
// B lane: col m, K=hi*2+{0,1} -> both are contiguous float2 loads from
// feature-major storage (no transpose needed). Then dist = sqi + sqq - 2G,
// diag +1e9, strict-insertion top-K per row (matches top_k tie semantics).
// ---------------------------------------------------------------------------
template <int K>
__global__ __launch_bounds__(256) void knn_wmma_kernel(const float* __restrict__ X,
                                                       int* __restrict__ idxOut) {
    __shared__ float sqAll[512];
    __shared__ float dist[16 * 512]; // 32KB
    const int b     = blockIdx.x >> 5;
    const int strip = blockIdx.x & 31;
    const int i0    = strip * 16;
    const float* Xb = X + (size_t)b * 512 * 64;

    const int tid  = threadIdx.x;
    const int wave = tid >> 5;
    const int lane = tid & 31;
    const int m    = lane & 15;
    const int hi   = lane >> 4;

    // squared norms of all points of this object
    for (int p = tid; p < 512; p += 256) {
        const float* xp = Xb + (size_t)p * 64;
        float s = 0.f;
        for (int c = 0; c < 64; ++c) s += xp[c] * xp[c];
        sqAll[p] = s;
    }
    __syncthreads();

    // gram strip: rows i0..i0+15, cols wave*64..wave*64+63
    const int q0w = wave * 64;
    float8 acc[4] = {};
    for (int k0 = 0; k0 < 64; k0 += 4) {
        f32x2 a = *(const f32x2*)(Xb + (size_t)(i0 + m) * 64 + k0 + hi * 2);
#pragma unroll
        for (int ct = 0; ct < 4; ++ct) {
            f32x2 bt = *(const f32x2*)(Xb + (size_t)(q0w + ct * 16 + m) * 64 + k0 + hi * 2);
            acc[ct] = __builtin_amdgcn_wmma_f32_16x16x4_f32(
                false, a, false, bt, (short)0, acc[ct], false, false);
        }
    }

    // dist = sq_i + sq_q - 2*G (+1e9 on diagonal); D lane: col n=ct*16+m,
    // VGPR r: row r+hi*8
#pragma unroll
    for (int ct = 0; ct < 4; ++ct) {
        int q = q0w + ct * 16 + m;
        float sqq = sqAll[q];
#pragma unroll
        for (int r = 0; r < 8; ++r) {
            int row = r + hi * 8;
            float d = sqAll[i0 + row] + sqq - 2.f * acc[ct][r];
            if (q == i0 + row) d += 1e9f;
            dist[row * 512 + q] = d;
        }
    }
    __syncthreads();

    // top-K per row
    if (tid < 16) {
        float best[K]; int bestI[K];
#pragma unroll
        for (int k = 0; k < K; ++k) { best[k] = 3.4e38f; bestI[k] = 0; }
        for (int q = 0; q < 512; ++q) {
            float d = dist[tid * 512 + q];
            if (d < best[K - 1]) {
                int k = K - 1;
                while (k > 0 && best[k - 1] > d) {
                    best[k] = best[k - 1]; bestI[k] = bestI[k - 1]; --k;
                }
                best[k] = d; bestI[k] = q;
            }
        }
        int* op = idxOut + ((size_t)b * 512 + i0 + tid) * K;
#pragma unroll
        for (int k = 0; k < K; ++k) op[k] = bestI[k];
    }
}

// ---------------------------------------------------------------------------
// Edge conv: one wave per point. E(16 x KPAD, fp16, rows=edges padded) @ Wt
// via v_wmma_f32_16x16x32_f16, then bias + LayerNorm + SELU + max over edges.
// KPAD = 32 (CIN=3) or 128 (CIN=64). Output channels fixed at 64.
// ---------------------------------------------------------------------------
template <int CIN, int KPAD, int KNB>
__global__ __launch_bounds__(256) void edgeconv_kernel(
    const float* __restrict__ X, const int* __restrict__ nbr,
    const _Float16* __restrict__ Wt, const float* __restrict__ bias,
    const float* __restrict__ gam, const float* __restrict__ bet,
    float* __restrict__ out, int P) {
    constexpr int KT = KPAD / 32;
    __shared__ __align__(16) _Float16 sW[64 * KPAD];             // transposed weights
    __shared__ __align__(16) unsigned char wbuf[8][16 * 64 * 4]; // E(f16) then D(f32)
    __shared__ float sXi[8][CIN];

    const int tid  = threadIdx.x;
    const int wave = tid >> 5;
    const int lane = tid & 31;
    const int m    = lane & 15;
    const int hi   = lane >> 4;

    for (int t = tid; t < 64 * KPAD; t += 256) sW[t] = Wt[t];

    const int pt = blockIdx.x * 8 + wave;
    const int b  = pt / P;
    const int i  = pt - b * P;
    const float* Xb = X + (size_t)b * P * CIN;
    const int*   nb = nbr + ((size_t)b * P + i) * KNB;

    for (int c = lane; c < CIN; c += 32) sXi[wave][c] = Xb[(size_t)i * CIN + c];
    __syncthreads();

    // Build edge tile E (16 x KPAD) in fp16: rows < KNB : [xj - xi | xi], rest 0
    _Float16* E = (_Float16*)wbuf[wave];
    for (int r = 0; r < 16; ++r) {
        int j = (r < KNB) ? nb[r] : 0;
        for (int c = lane; c < KPAD; c += 32) {
            float v = 0.f;
            if (r < KNB) {
                if (c < CIN)          v = Xb[(size_t)j * CIN + c] - sXi[wave][c];
                else if (c < 2 * CIN) v = sXi[wave][c - CIN];
            }
            E[r * KPAD + c] = (_Float16)v;
        }
    }
    lds_fence();

    // A operands: lane holds row m; K chunks at hi*8 and hi*8+16 within a 32-K tile
    half16 a[KT];
#pragma unroll
    for (int t = 0; t < KT; ++t) {
        const _Float16* p = E + m * KPAD + t * 32 + hi * 8;
        a[t] = load16h(p, p + 16);
    }

    // B operands from transposed weights: lane holds col n = ct*16+m,
    // K chunks at hi*8 and hi*8+16 within a 32-K tile
    float8 acc[4] = {};
#pragma unroll
    for (int ct = 0; ct < 4; ++ct) {
#pragma unroll
        for (int t = 0; t < KT; ++t) {
            const _Float16* p = sW + (ct * 16 + m) * KPAD + t * 32 + hi * 8;
            half16 bt = load16h(p, p + 16);
            acc[ct] = __builtin_amdgcn_wmma_f32_16x16x32_f16(
                false, a[t], false, bt, (short)0, acc[ct], false, false);
        }
    }

    // D layout: lane holds col n = ct*16 + m; VGPR r holds row m_r = r + hi*8
    float* D = (float*)wbuf[wave];
#pragma unroll
    for (int ct = 0; ct < 4; ++ct)
#pragma unroll
        for (int r = 0; r < 8; ++r)
            D[(r + hi * 8) * 64 + ct * 16 + m] = acc[ct][r];
    lds_fence();

    // bias + LayerNorm + SELU per edge row (fp32)
    if (lane < KNB) {
        float s1 = 0.f, s2 = 0.f;
        for (int n = 0; n < 64; ++n) {
            float y = D[lane * 64 + n] + bias[n];
            s1 += y; s2 += y * y;
        }
        float mean = s1 * (1.f / 64.f);
        float inv  = rsqrtf(s2 * (1.f / 64.f) - mean * mean + 1e-5f);
        for (int n = 0; n < 64; ++n) {
            float y = D[lane * 64 + n] + bias[n];
            float z = (y - mean) * inv * gam[n] + bet[n];
            D[lane * 64 + n] = selu_f(z);
        }
    }
    lds_fence();

    // max over edges
    float* Ob = out + ((size_t)b * P + i) * 64;
    for (int n = lane; n < 64; n += 32) {
        float mx = -3.4e38f;
        for (int r = 0; r < KNB; ++r) mx = fmaxf(mx, D[r * 64 + n]);
        Ob[n] = mx;
    }
}

// ---------------------------------------------------------------------------
// Max-pool over points: (NOBJ, P, 64) -> (NOBJ, 64)
// ---------------------------------------------------------------------------
__global__ __launch_bounds__(64) void pool_kernel(const float* __restrict__ in,
                                                  float* __restrict__ out, int P) {
    int b = blockIdx.x, c = threadIdx.x;
    float mx = -3.4e38f;
    for (int p = 0; p < P; ++p) mx = fmaxf(mx, in[((size_t)b * P + p) * 64 + c]);
    out[b * 64 + c] = mx;
}

// ---------------------------------------------------------------------------
// Cross-object attention: one block per sample (32 tokens x 64 dims), fp32
// ---------------------------------------------------------------------------
__global__ __launch_bounds__(256) void attn_kernel(
    const float* __restrict__ in, const float* __restrict__ Wq,
    const float* __restrict__ Wk, const float* __restrict__ Wv,
    const float* __restrict__ Wo, const float* __restrict__ bo,
    const float* __restrict__ g, const float* __restrict__ bet,
    float* __restrict__ out) {
    __shared__ float sx[2048], sq[2048], sk[2048], sv[2048], so[2048], sy[2048];
    __shared__ float sa[1024], mu[32], iv[32];
    const int smp = blockIdx.x, tid = threadIdx.x;
    const float* xin = in + smp * 2048;

    for (int t = tid; t < 2048; t += 256) sx[t] = xin[t];
    __syncthreads();
    for (int t = tid; t < 2048; t += 256) {
        int o = t >> 6, d = t & 63;
        float q = 0.f, k = 0.f, v = 0.f;
        for (int c = 0; c < 64; ++c) {
            float xv = sx[o * 64 + c];
            q += xv * Wq[c * 64 + d]; k += xv * Wk[c * 64 + d]; v += xv * Wv[c * 64 + d];
        }
        sq[t] = q; sk[t] = k; sv[t] = v;
    }
    __syncthreads();
    for (int t = tid; t < 1024; t += 256) {
        int o = t >> 5, p = t & 31;
        float acc = 0.f;
        for (int c = 0; c < 64; ++c) acc += sq[o * 64 + c] * sk[p * 64 + c];
        sa[t] = acc * 0.125f; // / sqrt(64)
    }
    __syncthreads();
    if (tid < 32) {
        float mx = -3.4e38f;
        for (int p = 0; p < 32; ++p) mx = fmaxf(mx, sa[tid * 32 + p]);
        float sum = 0.f;
        for (int p = 0; p < 32; ++p) { float e = expf(sa[tid * 32 + p] - mx); sa[tid * 32 + p] = e; sum += e; }
        float inv = 1.f / sum;
        for (int p = 0; p < 32; ++p) sa[tid * 32 + p] *= inv;
    }
    __syncthreads();
    for (int t = tid; t < 2048; t += 256) {
        int o = t >> 6, d = t & 63;
        float acc = 0.f;
        for (int p = 0; p < 32; ++p) acc += sa[o * 32 + p] * sv[p * 64 + d];
        so[t] = acc;
    }
    __syncthreads();
    for (int t = tid; t < 2048; t += 256) {
        int o = t >> 6, d = t & 63;
        float acc = bo[d];
        for (int c = 0; c < 64; ++c) acc += so[o * 64 + c] * Wo[c * 64 + d];
        sy[t] = acc;
    }
    __syncthreads();
    if (tid < 32) {
        float s1 = 0.f, s2 = 0.f;
        for (int d = 0; d < 64; ++d) { float v = sy[tid * 64 + d]; s1 += v; s2 += v * v; }
        float mean = s1 / 64.f;
        mu[tid] = mean; iv[tid] = rsqrtf(s2 / 64.f - mean * mean + 1e-5f);
    }
    __syncthreads();
    for (int t = tid; t < 2048; t += 256) {
        int o = t >> 6, d = t & 63;
        out[smp * 2048 + t] = selu_f((sy[t] - mu[o]) * iv[o] * g[d] + bet[d]);
    }
}

// ---------------------------------------------------------------------------
// bb_proj: [b1a | b2a] (64x128) @ W(128x64) + b -> LN -> SELU
// ---------------------------------------------------------------------------
__global__ __launch_bounds__(64) void bbproj_kernel(
    const float* __restrict__ b1, const float* __restrict__ b2,
    const float* __restrict__ W, const float* __restrict__ bias,
    const float* __restrict__ g, const float* __restrict__ bet,
    float* __restrict__ out) {
    __shared__ float row[128], y[64], mv[2];
    int r = blockIdx.x, n = threadIdx.x;
    for (int t = n; t < 128; t += 64)
        row[t] = (t < 64) ? b1[r * 64 + t] : b2[r * 64 + (t - 64)];
    __syncthreads();
    float acc = bias[n];
    for (int k = 0; k < 128; ++k) acc += row[k] * W[k * 64 + n];
    y[n] = acc;
    __syncthreads();
    if (n == 0) {
        float s1 = 0.f, s2 = 0.f;
        for (int k = 0; k < 64; ++k) { s1 += y[k]; s2 += y[k] * y[k]; }
        float mean = s1 / 64.f;
        mv[0] = mean; mv[1] = rsqrtf(s2 / 64.f - mean * mean + 1e-5f);
    }
    __syncthreads();
    out[r * 64 + n] = selu_f((y[n] - mv[0]) * mv[1] * g[n] + bet[n]);
}

// ---------------------------------------------------------------------------
// Final: [f1a f2a f3a f4a fb] (64x320) @ W(320x128) + b -> L2 normalize rows
// ---------------------------------------------------------------------------
__global__ __launch_bounds__(128) void final_kernel(
    const float* __restrict__ attnOut, const float* __restrict__ fb,
    const float* __restrict__ W, const float* __restrict__ bias,
    float* __restrict__ out) {
    __shared__ float row[320], y[128], nf;
    int r = blockIdx.x, n = threadIdx.x;
    for (int t = n; t < 320; t += 128) {
        int seg = t >> 6, c = t & 63;
        row[t] = (seg < 4) ? attnOut[seg * 4096 + r * 64 + c] : fb[r * 64 + c];
    }
    __syncthreads();
    float acc = bias[n];
    for (int k = 0; k < 320; ++k) acc += row[k] * W[k * 128 + n];
    y[n] = acc;
    __syncthreads();
    if (n == 0) {
        float s = 0.f;
        for (int k = 0; k < 128; ++k) s += y[k] * y[k];
        nf = 1.f / (sqrtf(s) + 1e-9f);
    }
    __syncthreads();
    out[r * 128 + n] = y[n] * nf;
}

// ---------------------------------------------------------------------------
// Host orchestration
// ---------------------------------------------------------------------------
extern "C" void kernel_launch(void* const* d_in, const int* in_sizes, int n_in,
                              void* d_out, int out_size, void* d_ws, size_t ws_size,
                              hipStream_t stream) {
    if (ws_size < (size_t)20 * 1024 * 1024) return;

    const float* x    = (const float*)d_in[0];
    const float* bbox = (const float*)d_in[1];

    // Detect params pytree flatten order: insertion (dict order) vs sorted keys.
    // insertion: in_sizes[16] = gc4.W = 8192 ; sorted: in_sizes[16] = bb_xo1.Wk = 4096
    bool sortedOrder = (n_in > 16 && in_sizes[16] == 4096);

    int gcBase[6], xoBase[6];
    int gcB, gcG, gcBe, xWq, xWk, xWv, xWo, xBo, xG, xBe, bbpBase, bbpB, bbpG, bbpBe, projBase;
    if (!sortedOrder) {
        int gb[6] = {4, 8, 12, 16, 48, 52};            // gc1..4, bb_gc1, bb_gc2
        int xb[6] = {20, 27, 34, 41, 56, 63};          // xo1..4, bb_xo1, bb_xo2
        for (int i = 0; i < 6; ++i) { gcBase[i] = gb[i]; xoBase[i] = xb[i]; }
        gcB = 1; gcG = 2; gcBe = 3;                    // W,b,g,beta
        xWq = 0; xWk = 1; xWv = 2; xWo = 3; xBo = 4; xG = 5; xBe = 6;
        bbpBase = 70; bbpB = 1; bbpG = 2; bbpBe = 3;
        projBase = 74;
    } else {
        int gb[6] = {30, 34, 38, 42, 4, 8};
        int xb[6] = {48, 55, 62, 69, 16, 23};
        for (int i = 0; i < 6; ++i) { gcBase[i] = gb[i]; xoBase[i] = xb[i]; }
        gcB = 1; gcBe = 2; gcG = 3;                    // W,b,beta,g
        xWk = 0; xWo = 1; xWq = 2; xWv = 3; xBo = 4; xBe = 5; xG = 6;
        bbpBase = 12; bbpB = 1; bbpBe = 2; bbpG = 3;
        projBase = 46;
    }
    auto F = [&](int i) { return (const float*)d_in[i]; };

    // Workspace layout (floats)
    float* wsf     = (float*)d_ws;
    float* featA   = wsf;                         // 64*512*64
    float* featB   = featA + 2097152;             // 64*512*64
    int*   knnIdx  = (int*)(featB + 2097152);     // 64*512*10
    float* pooled  = (float*)(knnIdx + 327680);   // 4 * 64*64
    float* attnOut = pooled + 16384;              // 4 * 64*64
    float* bA      = attnOut + 16384;             // 64*8*64
    float* bB      = bA + 32768;                  // 64*8*64
    int*   bIdx    = (int*)(bB + 32768);          // 64*8*4
    float* bPool   = (float*)(bIdx + 2048);       // 2 * 64*64
    float* bAttn   = bPool + 8192;                // 2 * 64*64
    float* fbP     = bAttn + 8192;                // 64*64
    _Float16* wt0  = (_Float16*)(fbP + 4096);     // gc1   : 64*32
    _Float16* wt1  = wt0 + 64 * 32;               // gc2   : 64*128
    _Float16* wt2  = wt1 + 64 * 128;              // gc3
    _Float16* wt3  = wt2 + 64 * 128;              // gc4
    _Float16* wt4  = wt3 + 64 * 128;              // bb_gc1: 64*32
    _Float16* wt5  = wt4 + 64 * 32;               // bb_gc2: 64*128

    // Weight conversion (fp32 -> transposed fp16, K padded)
    wcvt_kernel<<<dim3(8),  dim3(256), 0, stream>>>(F(gcBase[0]), 6,   wt0, 32);
    wcvt_kernel<<<dim3(32), dim3(256), 0, stream>>>(F(gcBase[1]), 128, wt1, 128);
    wcvt_kernel<<<dim3(32), dim3(256), 0, stream>>>(F(gcBase[2]), 128, wt2, 128);
    wcvt_kernel<<<dim3(32), dim3(256), 0, stream>>>(F(gcBase[3]), 128, wt3, 128);
    wcvt_kernel<<<dim3(8),  dim3(256), 0, stream>>>(F(gcBase[4]), 6,   wt4, 32);
    wcvt_kernel<<<dim3(32), dim3(256), 0, stream>>>(F(gcBase[5]), 128, wt5, 128);

    // ---- Point cloud branch: 4 graph convs ----
    knn_kernel<3, 10, 128><<<dim3(64), dim3(512), 0, stream>>>(x, knnIdx, 512);
    edgeconv_kernel<3, 32, 10><<<dim3(4096), dim3(256), 0, stream>>>(
        x, knnIdx, wt0, F(gcBase[0] + gcB), F(gcBase[0] + gcG), F(gcBase[0] + gcBe), featA, 512);
    pool_kernel<<<dim3(64), dim3(64), 0, stream>>>(featA, pooled + 0, 512);

    knn_wmma_kernel<10><<<dim3(2048), dim3(256), 0, stream>>>(featA, knnIdx);
    edgeconv_kernel<64, 128, 10><<<dim3(4096), dim3(256), 0, stream>>>(
        featA, knnIdx, wt1, F(gcBase[1] + gcB), F(gcBase[1] + gcG), F(gcBase[1] + gcBe), featB, 512);
    pool_kernel<<<dim3(64), dim3(64), 0, stream>>>(featB, pooled + 4096, 512);

    knn_wmma_kernel<10><<<dim3(2048), dim3(256), 0, stream>>>(featB, knnIdx);
    edgeconv_kernel<64, 128, 10><<<dim3(4096), dim3(256), 0, stream>>>(
        featB, knnIdx, wt2, F(gcBase[2] + gcB), F(gcBase[2] + gcG), F(gcBase[2] + gcBe), featA, 512);
    pool_kernel<<<dim3(64), dim3(64), 0, stream>>>(featA, pooled + 8192, 512);

    knn_wmma_kernel<10><<<dim3(2048), dim3(256), 0, stream>>>(featA, knnIdx);
    edgeconv_kernel<64, 128, 10><<<dim3(4096), dim3(256), 0, stream>>>(
        featA, knnIdx, wt3, F(gcBase[3] + gcB), F(gcBase[3] + gcG), F(gcBase[3] + gcBe), featB, 512);
    pool_kernel<<<dim3(64), dim3(64), 0, stream>>>(featB, pooled + 12288, 512);

    // Cross-object attention per layer
    for (int l = 0; l < 4; ++l) {
        int xb = xoBase[l];
        attn_kernel<<<dim3(2), dim3(256), 0, stream>>>(
            pooled + l * 4096, F(xb + xWq), F(xb + xWk), F(xb + xWv), F(xb + xWo),
            F(xb + xBo), F(xb + xG), F(xb + xBe), attnOut + l * 4096);
    }

    // ---- BBox branch: 2 graph convs ----
    knn_kernel<3, 4, 8><<<dim3(64), dim3(8), 0, stream>>>(bbox, bIdx, 8);
    edgeconv_kernel<3, 32, 4><<<dim3(64), dim3(256), 0, stream>>>(
        bbox, bIdx, wt4, F(gcBase[4] + gcB), F(gcBase[4] + gcG), F(gcBase[4] + gcBe), bA, 8);
    knn_kernel<64, 4, 8><<<dim3(64), dim3(8), 0, stream>>>(bA, bIdx, 8);
    edgeconv_kernel<64, 128, 4><<<dim3(64), dim3(256), 0, stream>>>(
        bA, bIdx, wt5, F(gcBase[5] + gcB), F(gcBase[5] + gcG), F(gcBase[5] + gcBe), bB, 8);
    pool_kernel<<<dim3(64), dim3(64), 0, stream>>>(bA, bPool, 8);
    pool_kernel<<<dim3(64), dim3(64), 0, stream>>>(bB, bPool + 4096, 8);

    for (int l = 0; l < 2; ++l) {
        int xb = xoBase[4 + l];
        attn_kernel<<<dim3(2), dim3(256), 0, stream>>>(
            bPool + l * 4096, F(xb + xWq), F(xb + xWk), F(xb + xWv), F(xb + xWo),
            F(xb + xBo), F(xb + xG), F(xb + xBe), bAttn + l * 4096);
    }

    bbproj_kernel<<<dim3(64), dim3(64), 0, stream>>>(
        bAttn, bAttn + 4096, F(bbpBase), F(bbpBase + bbpB), F(bbpBase + bbpG),
        F(bbpBase + bbpBe), fbP);

    // ---- Final projection + L2 normalize ----
    final_kernel<<<dim3(64), dim3(128), 0, stream>>>(
        attnOut, fbP, F(projBase), F(projBase + 1), (float*)d_out);
}
